// GraphConv_10892037062672
// MI455X (gfx1250) — matrix operand order repaired and checked by reading, not compiled, not run on hardware
//
#include <hip/hip_runtime.h>

typedef __attribute__((ext_vector_type(16))) _Float16 v16h;
typedef __attribute__((ext_vector_type(2)))  _Float16 v2h;
typedef __attribute__((ext_vector_type(8)))  float    v8f;
typedef __attribute__((ext_vector_type(4)))  unsigned int v4u;
typedef __attribute__((ext_vector_type(4)))  int v4i;

#define BB 4
#define NN 2048
#define FF 64
#define CC 64
#define LL 3
#define KSTEPS (NN / 32)

#define ROWSTRIDE 100   // 96 dwords of A-row data + 4 pad dwords (bank-conflict break)

#if defined(__gfx1250__) && __has_builtin(__builtin_amdgcn_global_load_async_to_lds_b128)
#define USE_ASYNC 1
#else
#define USE_ASYNC 0
#endif

// Builtin signature (from hipcc diagnostic): param0 = int4 in AS(1) (global),
// param1 = LDS side, then imm offset + cpol.
#define AS1V(p) ((__attribute__((address_space(1))) v4i*)(p))
#define AS3V(p) ((__attribute__((address_space(3))) v4i*)(p))

__device__ __forceinline__ void wait_async0() {
#if defined(__gfx1250__)
#if __has_builtin(__builtin_amdgcn_s_wait_asynccnt)
    __builtin_amdgcn_s_wait_asynccnt(0);
#else
    asm volatile("s_wait_asynccnt 0x0" ::: "memory");
#endif
#endif
}

// ---------------------------------------------------------------------------
// Stage 1: T_l[b,j,c] = sum_f h[l][c][f] * V[b][j][f]
//   l=0  -> Tself (f32, layout [b][j][c], same as output)
//   l=1..3 -> T (f16, layout [b][l-1][c][j]  -- c-major so WMMA B-fragments
//              are contiguous 32B per lane)
// ---------------------------------------------------------------------------
__global__ __launch_bounds__(256) void precompute_T(
    const float* __restrict__ V, const float* __restrict__ H,
    _Float16* __restrict__ T, float* __restrict__ Tself)
{
    int idx = blockIdx.x * 256 + threadIdx.x;      // BB*NN*CC threads
    int c = idx & 63;
    int j = (idx >> 6) & (NN - 1);
    int b = idx >> 17;                             // 17 = log2(NN*CC)

    const float* vr = V + ((size_t)b * NN + j) * FF;
    const float* h0 = H + ((size_t)0 * CC + c) * FF;
    const float* h1 = H + ((size_t)1 * CC + c) * FF;
    const float* h2 = H + ((size_t)2 * CC + c) * FF;
    const float* h3 = H + ((size_t)3 * CC + c) * FF;

    float s0 = 0.f, s1 = 0.f, s2 = 0.f, s3 = 0.f;
#pragma unroll
    for (int f = 0; f < FF; f += 4) {
        float4 vv = *(const float4*)(vr + f);
        float4 w;
        w = *(const float4*)(h0 + f);
        s0 = fmaf(vv.x, w.x, fmaf(vv.y, w.y, fmaf(vv.z, w.z, fmaf(vv.w, w.w, s0))));
        w = *(const float4*)(h1 + f);
        s1 = fmaf(vv.x, w.x, fmaf(vv.y, w.y, fmaf(vv.z, w.z, fmaf(vv.w, w.w, s1))));
        w = *(const float4*)(h2 + f);
        s2 = fmaf(vv.x, w.x, fmaf(vv.y, w.y, fmaf(vv.z, w.z, fmaf(vv.w, w.w, s2))));
        w = *(const float4*)(h3 + f);
        s3 = fmaf(vv.x, w.x, fmaf(vv.y, w.y, fmaf(vv.z, w.z, fmaf(vv.w, w.w, s3))));
    }

    Tself[((size_t)b * NN + j) * CC + c] = s0;
    T[(((size_t)b * LL + 0) * CC + c) * NN + j] = (_Float16)s1;
    T[(((size_t)b * LL + 1) * CC + c) * NN + j] = (_Float16)s2;
    T[(((size_t)b * LL + 2) * CC + c) * NN + j] = (_Float16)s3;
}

// ---------------------------------------------------------------------------
// Stage 2: out[b,i,c] = relu(Tself[b,i,c] + sum_l sum_j A[b,i,j,l]*T_l[b,j,c])
// One block = (b, 16-row i-tile); 4 waves, one 16-col c-tile per wave.
// A tile (16 x 32 x 3 f32 = 6KB) streamed into double-buffered LDS via
// global_load_async_to_lds_b128 (ASYNCcnt); 3 wmma f32_16x16x32_f16 / K-step.
// ---------------------------------------------------------------------------
__global__ __launch_bounds__(128) void graphconv_main(
    const float* __restrict__ A, const _Float16* __restrict__ T,
    const float* __restrict__ Tself, float* __restrict__ out)
{
    __shared__ float lds[2][16 * ROWSTRIDE];

    const int tid  = threadIdx.x;
    const int lane = tid & 31;
    const int wave = tid >> 5;
    const int b    = blockIdx.x >> 7;
    const int i0   = (blockIdx.x & 127) << 4;
    const int c0   = wave << 4;
    const int m    = lane & 15;          // row (A) / column (B) within tile
    const int hi   = lane >> 4;          // half-wave select
    const int ka   = hi << 4;            // A-fragment K offset: 0 or 16
    const int kb   = hi << 4;            // B-fragment K offset: 0 or 16

    const float* Ab = A + (size_t)b * NN * NN * LL;

    // Per-thread source/dest slots for the cooperative A-tile copy:
    // 384 b128 units per (16 x 32 x 3 f32) block, 3 per thread.
    const float* asrc[3];
    float*       adst[2][3];
#pragma unroll
    for (int t = 0; t < 3; ++t) {
        int u   = tid + (t << 7);        // b128 unit 0..383
        int row = u / 24;                // 24 b128 units per i-row
        int pos = u - row * 24;
        asrc[t]    = Ab + (size_t)(i0 + row) * NN * LL + (pos << 2);
        adst[0][t] = &lds[0][row * ROWSTRIDE + (pos << 2)];
        adst[1][t] = &lds[1][row * ROWSTRIDE + (pos << 2)];
    }

#if USE_ASYNC
    auto issueA = [&](int kt, int buf) {
        const int joff = kt * 96;        // 32 j * 3 l dwords per step
#pragma unroll
        for (int t = 0; t < 3; ++t) {
            __builtin_amdgcn_global_load_async_to_lds_b128(
                AS1V(asrc[t] + joff), AS3V(adst[buf][t]), 0, 0);
        }
    };
#else
    float4 pre[3];
    auto loadA = [&](int kt) {
        const int joff = kt * 96;
#pragma unroll
        for (int t = 0; t < 3; ++t) pre[t] = *(const float4*)(asrc[t] + joff);
    };
    auto stash = [&](int buf) {
#pragma unroll
        for (int t = 0; t < 3; ++t) *(float4*)adst[buf][t] = pre[t];
    };
#endif

    v8f acc = {};

    auto computeStep = [&](int kt, const float* S) {
        const int j0 = kt << 5;
#pragma unroll
        for (int l = 0; l < LL; ++l) {
            // B fragment: T_l[c0+m][j0+kb .. +15], contiguous f16 (32B)
            union { v4u u[2]; v16h h; } bf;
            const _Float16* tp =
                T + (((size_t)b * LL + l) * CC + (c0 + m)) * NN + j0 + kb;
            bf.u[0] = *(const v4u*)tp;
            bf.u[1] = *(const v4u*)(tp + 8);

            // A fragment: contiguous k = ka + s, same order as B, gather f32 from LDS, pack f16
            v16h af;
#pragma unroll
            for (int v = 0; v < 8; ++v) {
                int kk = (v << 1) + ka;
                float f0 = S[m * ROWSTRIDE + kk * LL + l];
                float f1 = S[m * ROWSTRIDE + (kk + 1) * LL + l];
                v2h p = __builtin_bit_cast(v2h,
                            __builtin_amdgcn_cvt_pkrtz(f0, f1));
                af[2 * v]     = p.x;
                af[2 * v + 1] = p.y;
            }

            acc = __builtin_amdgcn_wmma_f32_16x16x32_f16(
                false, af, false, bf.h, (short)0, acc, false, false);
        }
    };

#if USE_ASYNC
    issueA(0, 0);
    wait_async0();
    __syncthreads();
    for (int kt = 0; kt < KSTEPS; kt += 2) {
        issueA(kt + 1, 1);               // kt+1 <= KSTEPS-1 always here
        computeStep(kt, &lds[0][0]);
        wait_async0();
        __syncthreads();
        if (kt + 2 < KSTEPS) issueA(kt + 2, 0);
        computeStep(kt + 1, &lds[1][0]);
        wait_async0();
        __syncthreads();
    }
#else
    loadA(0);
    stash(0);
    __syncthreads();
    for (int kt = 0; kt < KSTEPS; kt += 2) {
        loadA(kt + 1);
        computeStep(kt, &lds[0][0]);
        stash(1);
        __syncthreads();
        if (kt + 2 < KSTEPS) loadA(kt + 2);
        computeStep(kt + 1, &lds[1][0]);
        if (kt + 2 < KSTEPS) stash(0);
        __syncthreads();
    }
#endif

    // Epilogue: add identity-edge term, relu, store (C/D VGPR layout:
    // VGPR r -> row i0 + r + 8*hi, column c0 + m)
    const size_t obase = ((size_t)b * NN + i0) * CC;
#pragma unroll
    for (int r = 0; r < 8; ++r) {
        int row = r + (hi << 3);
        size_t off = obase + (size_t)row * CC + c0 + m;
        float val = acc[r] + Tself[off];
        out[off] = val > 0.f ? val : 0.f;
    }
}

// ---------------------------------------------------------------------------
extern "C" void kernel_launch(void* const* d_in, const int* in_sizes, int n_in,
                              void* d_out, int out_size, void* d_ws, size_t ws_size,
                              hipStream_t stream) {
    (void)in_sizes; (void)n_in; (void)out_size; (void)ws_size;

    const float* V = (const float*)d_in[0];
    const float* A = (const float*)d_in[1];
    const float* H = (const float*)d_in[2];
    float* out = (float*)d_out;

    // Workspace: T (f16, 4*3*64*2048 = 3 MB) then Tself (f32, 2 MB)
    _Float16* T = (_Float16*)d_ws;
    float* Tself = (float*)((char*)d_ws +
                            (size_t)BB * LL * CC * NN * sizeof(_Float16));

    precompute_T<<<(BB * NN * CC) / 256, 256, 0, stream>>>(V, H, T, Tself);
    graphconv_main<<<BB * (NN / 16), 128, 0, stream>>>(A, T, Tself, out);
}
